// KANLayer_5652176962254
// MI455X (gfx1250) — matrix operand order repaired and checked by reading, not compiled, not run on hardware
//
#include <hip/hip_runtime.h>

typedef __attribute__((ext_vector_type(16))) _Float16 v16h;
typedef __attribute__((ext_vector_type(8)))  float    v8f;

#define IN_F   512
#define NB     8
#define OUT_F  512
#define NROWS  8192
#define K1     4096                 // phi region of the fused K dimension
#define KTOT   4608                 // + cos(x) @ scale_base region
#define BK     32
#define STAGES (KTOT / BK)          // 144; K1/BK = 128 exactly -> uniform branches
#define BM     128
#define BN     128
#define BETA   (64.0f / 9.0f)       // (nb/log2(nb))^2 = (8/3)^2

__global__ __launch_bounds__(256, 1)
void kan_wmma_kernel(const float* __restrict__ x,
                     const float* __restrict__ rw,
                     const float* __restrict__ rc,
                     const float* __restrict__ weight,
                     const float* __restrict__ bias,
                     const float* __restrict__ scale_base,
                     float* __restrict__ out)
{
    // LDS tiles stored directly in WMMA fragment order:
    // [buf][frag][lane][dword]  (16 halves per lane per fragment = 32B contiguous)
    __shared__ __align__(32) unsigned int ldsA[2][8][32][8];   // 16 KB
    __shared__ __align__(32) unsigned int ldsB[2][8][32][8];   // 16 KB

    const int tid  = threadIdx.x;
    const int lane = tid & 31;
    const int wid  = tid >> 5;
    const int wm   = wid & 3;        // 4 waves along M (4*32 = 128 rows)
    const int wn   = wid >> 2;       // 2 waves along N (2*64 = 128 cols)
    const int rowBase = blockIdx.x * BM;
    const int colBase = blockIdx.y * BN;

    v8f acc[2][4] = {};              // 8 accum fragments = 64 VGPRs

    uint4        regA[2];            // 2 tasks x 8 halves (one x-element's 8 bases)
    unsigned int regB[8];

    // ---- produce next stage into registers (global loads + phi/cos math) ----
    auto loadStage = [&](int s) {
        const int kBase = s * BK;
        // A: task = (row, il); covers k = 8*il .. 8*il+7 of this stage
        #pragma unroll
        for (int p = 0; p < 2; ++p) {
            const int idx = p * 256 + tid;          // 0..511
            const int row = idx >> 2;               // 0..127
            const int il  = idx & 3;                // 0..3
            float h[8];
            if (kBase < K1) {                       // phi region: 8 bases of one x
                const int i = (kBase >> 3) + il;
                const float xv = x[(size_t)(rowBase + row) * IN_F + i];
                float wv[8], cv[8];
                *(float4*)&wv[0] = *(const float4*)(rw + i * NB);
                *(float4*)&wv[4] = *(const float4*)(rw + i * NB + 4);
                *(float4*)&cv[0] = *(const float4*)(rc + i * NB);
                *(float4*)&cv[4] = *(const float4*)(rc + i * NB + 4);
                #pragma unroll
                for (int j = 0; j < 8; ++j) {
                    const float z = xv * wv[j] - cv[j];
                    h[j] = __expf(-BETA * z * z);
                }
            } else {                                // cos region: 8 consecutive x
                const int i = (kBase - K1) + 8 * il;
                const float* xp = x + (size_t)(rowBase + row) * IN_F + i;
                float xv[8];
                *(float4*)&xv[0] = *(const float4*)(xp);
                *(float4*)&xv[4] = *(const float4*)(xp + 4);
                #pragma unroll
                for (int j = 0; j < 8; ++j) h[j] = __cosf(xv[j]);
            }
            union { _Float16 hh[8]; uint4 u; } pa;
            #pragma unroll
            for (int j = 0; j < 8; ++j) pa.hh[j] = (_Float16)h[j];
            regA[p] = pa.u;
        }
        // B: pair (k,k+1) for one col; coalesced over col across threads
        #pragma unroll
        for (int p = 0; p < 8; ++p) {
            const int idx = p * 256 + tid;          // 0..2047
            const int col = idx & 127;
            const int kb  = (idx >> 7) * 2;         // 0,2,...,30
            const int gkb = kBase + kb;
            float b0, b1;
            if (gkb < K1) {
                b0 = weight[(size_t)gkb       * OUT_F + colBase + col];
                b1 = weight[(size_t)(gkb + 1) * OUT_F + colBase + col];
            } else {
                const int r = gkb - K1;
                b0 = scale_base[(size_t)r       * OUT_F + colBase + col];
                b1 = scale_base[(size_t)(r + 1) * OUT_F + colBase + col];
            }
            union { _Float16 h[2]; unsigned int u; } pb;
            pb.h[0] = (_Float16)b0; pb.h[1] = (_Float16)b1;
            regB[p] = pb.u;
        }
    };

    // ---- scatter staged registers into LDS in WMMA fragment layout ----
    auto storeStage = [&](int buf) {
        // A (16-bit 16x32: lane group = k bit4 = il>>1; 8 halves -> 4 dwords contiguous)
        #pragma unroll
        for (int p = 0; p < 2; ++p) {
            const int idx = p * 256 + tid;
            const int row = idx >> 2;
            const int il  = idx & 3;
            const int mf  = row >> 4, rr = row & 15;
            const int lA  = rr | ((il >> 1) << 4);
            const int dw  = (il & 1) * 4;           // slot base 0 or 8 halves
            *(uint4*)&ldsA[buf][mf][lA][dw] = regA[p];
        }
        // B (16-bit 32x16: lane group = k bit4, slot = k&15)
        #pragma unroll
        for (int p = 0; p < 8; ++p) {
            const int idx = p * 256 + tid;
            const int col = idx & 127;
            const int kb  = (idx >> 7) * 2;
            const int nf  = col >> 4, cc = col & 15;
            const int lB  = cc | (((kb >> 4) & 1) << 4);
            ldsB[buf][nf][lB][(kb & 15) >> 1] = regB[p];
        }
    };

    loadStage(0);
    storeStage(0);
    __syncthreads();

    for (int s = 0; s < STAGES; ++s) {
        const int buf = s & 1;
        if (s + 1 < STAGES) loadStage(s + 1);     // global loads for next stage

        v16h a[2], b[4];
        #pragma unroll
        for (int mi = 0; mi < 2; ++mi)
            a[mi] = *(const v16h*)&ldsA[buf][wm * 2 + mi][lane][0];
        #pragma unroll
        for (int ni = 0; ni < 4; ++ni)
            b[ni] = *(const v16h*)&ldsB[buf][wn * 4 + ni][lane][0];

        #pragma unroll
        for (int mi = 0; mi < 2; ++mi)
            #pragma unroll
            for (int ni = 0; ni < 4; ++ni)
                acc[mi][ni] = __builtin_amdgcn_wmma_f32_16x16x32_f16(
                    false, a[mi], false, b[ni],
                    (short)0, acc[mi][ni], false, false);

        if (s + 1 < STAGES) storeStage(buf ^ 1);  // fill the other buffer
        __syncthreads();
    }

    // ---- epilogue: += bias, store (C layout: VGPR r -> M=r / M=r+8 by lane half)
    const int rOff = (lane >> 4) * 8;
    #pragma unroll
    for (int ni = 0; ni < 4; ++ni) {
        const int col = colBase + (wn * 4 + ni) * 16 + (lane & 15);
        const float bv = bias[col];
        #pragma unroll
        for (int mi = 0; mi < 2; ++mi) {
            const int m0 = rowBase + (wm * 2 + mi) * 16 + rOff;
            #pragma unroll
            for (int r = 0; r < 8; ++r)
                out[(size_t)(m0 + r) * OUT_F + col] = acc[mi][ni][r] + bv;
        }
    }
}

extern "C" void kernel_launch(void* const* d_in, const int* in_sizes, int n_in,
                              void* d_out, int out_size, void* d_ws, size_t ws_size,
                              hipStream_t stream) {
    const float* x    = (const float*)d_in[0];
    const float* rw   = (const float*)d_in[1];
    const float* rc   = (const float*)d_in[2];
    const float* w    = (const float*)d_in[3];
    const float* bias = (const float*)d_in[4];
    const float* sb   = (const float*)d_in[5];
    float* out = (float*)d_out;

    dim3 grid(NROWS / BM, OUT_F / BN);   // 64 x 4 = 256 workgroups
    kan_wmma_kernel<<<grid, 256, 0, stream>>>(x, rw, rc, w, bias, sb, out);
}
